// LNNAttentionLayer_78211354460418
// MI455X (gfx1250) — matrix operand order repaired and checked by reading, not compiled, not run on hardware
//
#include <hip/hip_runtime.h>
#include <math.h>

// Problem constants (reference: B,S,D,H = 2,2048,1024,16; HD=64)
#define BB   2
#define SS   2048
#define DDIM 1024
#define NH   16
#define HD   64

typedef __attribute__((ext_vector_type(16))) __bf16 v16bf;
typedef __attribute__((ext_vector_type(8)))  float  v8f;
typedef __attribute__((ext_vector_type(8)))  __bf16 bf16x8;
typedef __attribute__((ext_vector_type(4)))  __bf16 bf16x4;

union Frag { v16bf v; bf16x8 h[2]; };

__device__ __forceinline__ v8f wmma_bf16(const Frag& a, const Frag& b, v8f c) {
    // D = A(16x32 bf16) x B(32x16 bf16) + C(16x16 f32)
    return __builtin_amdgcn_wmma_f32_16x16x32_bf16(
        false, a.v, false, b.v, (short)0, c, false, false);
}

// single v_rcp_f32 instead of the IEEE div_scale/div_fmas/div_fixup sequence
__device__ __forceinline__ float fast_rcp(float x) {
    return __builtin_amdgcn_rcpf(x);
}

// -------------------------------------------------------------------------
// fp32 -> bf16 bulk convert (one-time; amortizes conversion out of the GEMMs)
// -------------------------------------------------------------------------
__global__ __launch_bounds__(256)
void cvt_kernel(const float* __restrict__ src, __bf16* __restrict__ dst, int n4) {
    const int idx = blockIdx.x * 256 + threadIdx.x;   // float4 granules
    if (idx < n4) {
        float4 v = ((const float4*)src)[idx];
        bf16x4 o = { (__bf16)v.x, (__bf16)v.y, (__bf16)v.z, (__bf16)v.w };
        ((bf16x4*)dst)[idx] = o;
    }
}

// -------------------------------------------------------------------------
// Small projections: li = x @ relW[:, :D].T ; lj = x @ relW[:, D:].T ;
// gate = sigmoid(x @ gateW.T + gateB).  One block per (b,s) row.
// -------------------------------------------------------------------------
__global__ __launch_bounds__(128)
void proj_small_kernel(const float* __restrict__ x,
                       const float* __restrict__ relW,
                       const float* __restrict__ gateW,
                       const float* __restrict__ gateB,
                       float* __restrict__ li, float* __restrict__ lj,
                       float* __restrict__ gate) {
    __shared__ float xs[DDIM];
    const int row = blockIdx.x;                 // b*S + s
    const float* xr = x + (size_t)row * DDIM;
    for (int k = threadIdx.x; k < DDIM; k += 128) xs[k] = xr[k];
    __syncthreads();

    const int lane = threadIdx.x & 31;
    const int wid  = threadIdx.x >> 5;
    for (int o = wid; o < 3 + 3 + NH; o += 4) {
        const float* wrow;
        float bsc = 0.f;
        int kind;
        if (o < 3)      { wrow = relW  + (size_t)o * (2 * DDIM);              kind = 0; }
        else if (o < 6) { wrow = relW  + (size_t)(o - 3) * (2 * DDIM) + DDIM; kind = 1; }
        else            { wrow = gateW + (size_t)(o - 6) * DDIM; bsc = gateB[o - 6]; kind = 2; }
        float sum = 0.f;
        for (int k = lane; k < DDIM; k += 32) sum += xs[k] * wrow[k];
        #pragma unroll
        for (int mm = 16; mm >= 1; mm >>= 1) sum += __shfl_xor(sum, mm, 32);
        if (lane == 0) {
            if (kind == 0)      li[(size_t)row * 3 + o]       = sum;
            else if (kind == 1) lj[(size_t)row * 3 + (o - 3)] = sum;
            else gate[(size_t)row * NH + (o - 6)] = fast_rcp(1.f + __expf(-(sum + bsc)));
        }
    }
}

// -------------------------------------------------------------------------
// Barrier-free WMMA GEMM on pre-converted bf16 operands.
// Y = A(M x 1024) @ W^T + bias.  W bf16 row-major (N x K) == B-frag layout;
// A bf16 row-major (M x K) == A-frag layout, so fragments are direct
// pairs of 16-byte global loads, 16 elements apart.  Wave tile 16(M) x 128(N): 8 WMMAs per
// K-step, no LDS, no barriers; reuse is served by L0/L2 (all operands fit L2).
// MODE: 0 -> Q (b,h,s,hd) bf16 ; 1 -> K (b,h,s,hd) bf16 ;
//       2 -> V transposed (b,h,hd,s) bf16 ; 3 -> fp32 (M x N) output.
// -------------------------------------------------------------------------
template<int MODE>
__global__ __launch_bounds__(256)
void gemm_bf16(const __bf16* __restrict__ A, const __bf16* __restrict__ Wb,
               const float* __restrict__ bias, void* __restrict__ Out) {
    const int K      = DDIM;
    const int lane   = threadIdx.x & 31;
    const int wid    = threadIdx.x >> 5;
    const int lane15 = lane & 15;
    const int hh     = lane >> 4;
    const int gw     = blockIdx.x * 8 + wid;    // global wave id
    const int m0     = (gw >> 3) * 16;          // 256 M tiles
    const int n0     = (gw & 7) * 128;          // 8 N groups of 128

    const __bf16* arow = A + (size_t)(m0 + lane15) * K;
    const int c0a = hh ? 8 : 0;
    const int c0b = hh ? 8 : 0;

    v8f acc[8] = {};
    for (int k0 = 0; k0 < K; k0 += 32) {
        if (k0 + 64 < K) __builtin_prefetch((const void*)(arow + k0 + 64), 0, 1);
        Frag a;
        a.h[0] = *(const bf16x8*)(arow + k0 + c0a);
        a.h[1] = *(const bf16x8*)(arow + k0 + c0a + 16);
        #pragma unroll
        for (int nt = 0; nt < 8; ++nt) {
            const __bf16* brow = Wb + (size_t)(n0 + nt * 16 + lane15) * K + k0 + c0b;
            Frag bfr;
            bfr.h[0] = *(const bf16x8*)brow;
            bfr.h[1] = *(const bf16x8*)(brow + 16);
            acc[nt] = wmma_bf16(a, bfr, acc[nt]);
        }
    }

    // epilogue: C layout rows r + 8*hh, col = lane15
    #pragma unroll
    for (int nt = 0; nt < 8; ++nt) {
        const int gn = n0 + nt * 16 + lane15;
        const float bv = bias[gn];
        #pragma unroll
        for (int r = 0; r < 8; ++r) {
            const int gm = m0 + r + hh * 8;
            const float val = acc[nt][r] + bv;
            if (MODE == 3) {
                ((float*)Out)[(size_t)gm * DDIM + gn] = val;
            } else {
                const int bidx = gm / SS, s = gm % SS;
                const int hidx = gn >> 6, hd = gn & 63;
                __bf16* O = (__bf16*)Out;
                if (MODE == 2)
                    O[(((size_t)bidx * NH + hidx) * HD + hd) * SS + s] = (__bf16)val;
                else
                    O[(((size_t)bidx * NH + hidx) * SS + s) * HD + hd] = (__bf16)val;
            }
        }
    }
}

// -------------------------------------------------------------------------
// Flash attention, one block per (b, 16-row i-tile) covering ALL 16 heads
// (16 waves, 512 threads).  The (i,j) relation-softmax probs are computed
// cooperatively ONCE per block into LDS and shared by every head wave.
// 3 block-wide barriers per 32-wide j-chunk.
// -------------------------------------------------------------------------
__global__ __launch_bounds__(512)
void attn_kernel(const __bf16* __restrict__ Q, const __bf16* __restrict__ Kk,
                 const __bf16* __restrict__ Vt,
                 const float* __restrict__ LI, const float* __restrict__ LJ,
                 const float* __restrict__ GATE,
                 const float* __restrict__ relb, const float* __restrict__ LB,
                 __bf16* __restrict__ Ctx) {
    __shared__ float  eliS[16][4];        // exp(li_c + rel_b_c) per row
    __shared__ float  probS[16][32][4];   // normalized 3-class probs
    __shared__ __bf16 Plds[NH][16][40];   // per-wave P transpose buffer

    const int tid    = threadIdx.x;
    const int lane   = tid & 31;
    const int wid    = tid >> 5;          // head index
    const int lane15 = lane & 15;
    const int hh     = lane >> 4;

    const int b  = blockIdx.x / (SS / 16);
    const int i0 = (blockIdx.x % (SS / 16)) * 16;
    const int h  = wid;

    const size_t bh = (size_t)b * NH + h;
    const __bf16* Qp = Q  + (bh * SS + i0) * HD;
    const __bf16* Kp = Kk + bh * SS * HD;
    const __bf16* Vp = Vt + bh * HD * SS;

    // exp(li + rel_b) for the 16 rows -> LDS (once per block)
    if (tid < 48) {
        const int row = tid / 3, c = tid % 3;
        eliS[row][c] = __expf(LI[((size_t)b * SS + i0 + row) * 3 + c] + relb[c]);
    }

    // Q fragments (16x64 -> two K=32 A-fragments), straight from global
    Frag qa[2];
    {
        const int c0 = hh ? 8 : 0;
        #pragma unroll
        for (int f = 0; f < 2; ++f) {
            qa[f].h[0] = *(const bf16x8*)(Qp + (size_t)lane15 * HD + f * 32 + c0);
            qa[f].h[1] = *(const bf16x8*)(Qp + (size_t)lane15 * HD + f * 32 + c0 + 16);
        }
    }
    // gate per row for this head
    float g[8];
    #pragma unroll
    for (int r = 0; r < 8; ++r)
        g[r] = GATE[((size_t)b * SS + i0 + r + hh * 8) * NH + h];
    const float lb0 = LB[h * 3 + 0], lb1 = LB[h * 3 + 1], lb2 = LB[h * 3 + 2];

    v8f acc[4] = {};
    float mrun[8], lrun[8];
    #pragma unroll
    for (int r = 0; r < 8; ++r) { mrun[r] = -1e30f; lrun[r] = 0.f; }
    const float iscale = 0.125f;   // 1/sqrt(64)

    __syncthreads();   // eliS visible before first probs phase

    for (int j0 = 0; j0 < SS; j0 += 32) {
        // phase 1: normalized relation probs, one (row,j) element per thread;
        // exp(lj) computed inline (3 trans ops) -- cheaper than a barrier.
        {
            const int row = tid >> 5, j = tid & 31;
            const size_t jb = (size_t)b * SS + j0 + j;
            const float w0 = eliS[row][0] * __expf(LJ[jb * 3 + 0]);
            const float w1 = eliS[row][1] * __expf(LJ[jb * 3 + 1]);
            const float w2 = eliS[row][2] * __expf(LJ[jb * 3 + 2]);
            const float inv = fast_rcp(w0 + w1 + w2);
            probS[row][j][0] = w0 * inv;
            probS[row][j][1] = w1 * inv;
            probS[row][j][2] = w2 * inv;
        }
        // QK^T WMMAs (independent of probs -> overlaps the shared VALU work)
        v8f sv[2];
        #pragma unroll
        for (int sub = 0; sub < 2; ++sub) {
            const int jn = j0 + sub * 16 + lane15;
            const int ck = hh ? 8 : 0;
            const __bf16* kr = Kp + (size_t)jn * HD;
            Frag kb0, kb1;
            kb0.h[0] = *(const bf16x8*)(kr + ck);
            kb0.h[1] = *(const bf16x8*)(kr + ck + 16);
            kb1.h[0] = *(const bf16x8*)(kr + 32 + ck);
            kb1.h[1] = *(const bf16x8*)(kr + 32 + ck + 16);
            v8f s = {};
            s = wmma_bf16(qa[0], kb0, s);
            s = wmma_bf16(qa[1], kb1, s);
            sv[sub] = s;
        }
        __syncthreads();
        // phase 2: bias (3 LDS reads + 3 FMA per element) + online softmax
        float pv[2][8];
        #pragma unroll
        for (int sub = 0; sub < 2; ++sub) {
            const int jl = sub * 16 + lane15;
            #pragma unroll
            for (int r = 0; r < 8; ++r) {
                const int row = r + hh * 8;
                const float biasv = probS[row][jl][0] * lb0 +
                                    probS[row][jl][1] * lb1 +
                                    probS[row][jl][2] * lb2;
                pv[sub][r] = sv[sub][r] * iscale + g[r] * biasv;
            }
        }
        float alpha[8];
        #pragma unroll
        for (int r = 0; r < 8; ++r) {
            float tmax = fmaxf(pv[0][r], pv[1][r]);
            #pragma unroll
            for (int mm = 8; mm >= 1; mm >>= 1)
                tmax = fmaxf(tmax, __shfl_xor(tmax, mm, 32));
            const float mnew = fmaxf(mrun[r], tmax);
            alpha[r] = __expf(mrun[r] - mnew);
            mrun[r]  = mnew;
            pv[0][r] = __expf(pv[0][r] - mnew);
            pv[1][r] = __expf(pv[1][r] - mnew);
            float rs = pv[0][r] + pv[1][r];
            #pragma unroll
            for (int mm = 8; mm >= 1; mm >>= 1) rs += __shfl_xor(rs, mm, 32);
            lrun[r] = lrun[r] * alpha[r] + rs;
        }
        #pragma unroll
        for (int nt = 0; nt < 4; ++nt)
            #pragma unroll
            for (int r = 0; r < 8; ++r) acc[nt][r] *= alpha[r];

        // phase 3: transpose P (C layout -> A-frag layout) via per-wave LDS
        #pragma unroll
        for (int sub = 0; sub < 2; ++sub)
            #pragma unroll
            for (int r = 0; r < 8; ++r)
                Plds[wid][r + hh * 8][sub * 16 + lane15] = (__bf16)pv[sub][r];
        __syncthreads();
        Frag pa;
        {
            const int c0 = hh ? 8 : 0;
            pa.h[0] = *(const bf16x8*)&Plds[wid][lane15][c0];
            pa.h[1] = *(const bf16x8*)&Plds[wid][lane15][c0 + 16];
        }
        // P(16x32) @ V(32x64): Vt rows contiguous in j -> direct B-frags
        #pragma unroll
        for (int nt = 0; nt < 4; ++nt) {
            const int ck = hh ? 8 : 0;
            const __bf16* vr = Vp + (size_t)(nt * 16 + lane15) * SS + j0 + ck;
            Frag vb;
            vb.h[0] = *(const bf16x8*)vr;
            vb.h[1] = *(const bf16x8*)(vr + 16);
            acc[nt] = wmma_bf16(pa, vb, acc[nt]);
        }
        __syncthreads();
    }

    // normalize (v_rcp, not IEEE div) and write context in (b,s,d) bf16
    float invl[8];
    #pragma unroll
    for (int r = 0; r < 8; ++r) invl[r] = fast_rcp(lrun[r]);
    #pragma unroll
    for (int nt = 0; nt < 4; ++nt) {
        #pragma unroll
        for (int r = 0; r < 8; ++r) {
            const int i   = i0 + r + hh * 8;
            const int col = h * HD + nt * 16 + lane15;
            Ctx[((size_t)b * SS + i) * DDIM + col] = (__bf16)(acc[nt][r] * invl[r]);
        }
    }
}

// -------------------------------------------------------------------------
extern "C" void kernel_launch(void* const* d_in, const int* in_sizes, int n_in,
                              void* d_out, int out_size, void* d_ws, size_t ws_size,
                              hipStream_t stream) {
    (void)in_sizes; (void)n_in; (void)out_size; (void)ws_size;
    const float* x    = (const float*)d_in[0];
    const float* Wq   = (const float*)d_in[1];
    const float* bq   = (const float*)d_in[2];
    const float* Wk   = (const float*)d_in[3];
    const float* bk   = (const float*)d_in[4];
    const float* Wv   = (const float*)d_in[5];
    const float* bv   = (const float*)d_in[6];
    const float* Wo   = (const float*)d_in[7];
    const float* bo   = (const float*)d_in[8];
    const float* relW = (const float*)d_in[9];
    const float* relb = (const float*)d_in[10];
    const float* LB   = (const float*)d_in[11];
    const float* gW   = (const float*)d_in[12];
    const float* gB   = (const float*)d_in[13];
    float* out = (float*)d_out;

    char* ws = (char*)d_ws;
    const size_t QE = (size_t)BB * NH * SS * HD;        // elements per Q/K/V
    const size_t XE = (size_t)BB * SS * DDIM;           // elements of x / ctx
    const size_t WE = (size_t)DDIM * DDIM;              // elements per weight
    __bf16* Qw  = (__bf16*)ws;  ws += QE * 2;
    __bf16* Kw  = (__bf16*)ws;  ws += QE * 2;
    __bf16* Vtw = (__bf16*)ws;  ws += QE * 2;
    __bf16* Ctx = (__bf16*)ws;  ws += XE * 2;
    __bf16* Xb  = (__bf16*)ws;  ws += XE * 2;
    __bf16* Wqb = (__bf16*)ws;  ws += WE * 2;
    __bf16* Wkb = (__bf16*)ws;  ws += WE * 2;
    __bf16* Wvb = (__bf16*)ws;  ws += WE * 2;
    __bf16* Wob = (__bf16*)ws;  ws += WE * 2;
    float*  li  = (float*)ws;   ws += (size_t)BB * SS * 3 * 4;
    float*  lj  = (float*)ws;   ws += (size_t)BB * SS * 3 * 4;
    float*  gt  = (float*)ws;   ws += (size_t)BB * SS * NH * 4;

    // one-time bf16 conversions (pure bandwidth; tiny vs 23.3 TB/s)
    cvt_kernel<<<(int)(XE / 4 + 255) / 256, 256, 0, stream>>>(x,  Xb,  (int)(XE / 4));
    cvt_kernel<<<(int)(WE / 4 + 255) / 256, 256, 0, stream>>>(Wq, Wqb, (int)(WE / 4));
    cvt_kernel<<<(int)(WE / 4 + 255) / 256, 256, 0, stream>>>(Wk, Wkb, (int)(WE / 4));
    cvt_kernel<<<(int)(WE / 4 + 255) / 256, 256, 0, stream>>>(Wv, Wvb, (int)(WE / 4));
    cvt_kernel<<<(int)(WE / 4 + 255) / 256, 256, 0, stream>>>(Wo, Wob, (int)(WE / 4));

    proj_small_kernel<<<BB * SS, 128, 0, stream>>>(x, relW, gW, gB, li, lj, gt);

    const int gemm_blocks = ((BB * SS) / 16) * (DDIM / 128) / 8;   // 256
    gemm_bf16<0><<<gemm_blocks, 256, 0, stream>>>(Xb, Wqb, bq, Qw);
    gemm_bf16<1><<<gemm_blocks, 256, 0, stream>>>(Xb, Wkb, bk, Kw);
    gemm_bf16<2><<<gemm_blocks, 256, 0, stream>>>(Xb, Wvb, bv, Vtw);

    attn_kernel<<<BB * (SS / 16), 512, 0, stream>>>(Qw, Kw, Vtw, li, lj, gt, relb, LB, Ctx);

    gemm_bf16<3><<<gemm_blocks, 256, 0, stream>>>(Ctx, Wob, bo, out);
}
